// PaTHAttention_88476326298197
// MI455X (gfx1250) — matrix operand order repaired and checked by reading, not compiled, not run on hardware
//
#include <hip/hip_runtime.h>
#include <hip/hip_bf16.h>

// ---------------------------------------------------------------------------
// PaTH attention forward for MI455X (gfx1250): all GEMM-shaped work on
// v_wmma_f32_16x16x32_bf16 (wave32, 16x16 tiles, f32 accumulate).
// ---------------------------------------------------------------------------

#define TDIM   1024
#define DMODEL 1024
#define HDh    64
#define HRh    32
#define OUTWc  4096
#define SCALEc 0.125f   // 64^-0.5

typedef unsigned short u16;
typedef __attribute__((ext_vector_type(16))) __bf16 v16bf;
typedef __attribute__((ext_vector_type(8)))  float  v8f;

#define ZERO8 {0.f,0.f,0.f,0.f,0.f,0.f,0.f,0.f}
#define WMMA(a, b, c) \
  __builtin_amdgcn_wmma_f32_16x16x32_bf16(false, (a), false, (b), (short)0, (c), false, false)

__device__ __forceinline__ u16 f2bf(float f) {
  union { float f; unsigned int u; } x; x.f = f;
  unsigned int u = x.u;
  u += 0x7FFFu + ((u >> 16) & 1u);           // round-to-nearest-even
  return (u16)(u >> 16);
}
__device__ __forceinline__ float bf2f(u16 b) {
  union { unsigned int u; float f; } x; x.u = ((unsigned int)b) << 16;
  return x.f;
}

union FragU { float4 q[2]; v16bf v; };

// A fragment (M x K row-major source): lane holds row (lane&15).
// ISA layout: e=0..7 -> K = k0 + 8*(lane>>4) + e ; e=8..15 -> +16.
__device__ __forceinline__ v16bf frag_a(const u16* base, int ld, int row, int k0) {
  const int lane = threadIdx.x & 31;
  const u16* p = base + (size_t)(row + (lane & 15)) * (size_t)ld
                      + (size_t)(k0 + ((lane >> 4) << 3));
  FragU u;
  u.q[0] = *reinterpret_cast<const float4*>(p);
  u.q[1] = *reinterpret_cast<const float4*>(p + 16);
  return u.v;
}
// B fragment for C = A * B^T (N x K row-major source): lane holds col (lane&15),
// same layout as A: e=0..7 -> K = k0 + 8*(lane>>4) + e ; e=8..15 -> +16.
__device__ __forceinline__ v16bf frag_b(const u16* base, int ld, int col, int k0) {
  const int lane = threadIdx.x & 31;
  const u16* p = base + (size_t)(col + (lane & 15)) * (size_t)ld
                      + (size_t)(k0 + ((lane >> 4) << 3));
  FragU u;
  u.q[0] = *reinterpret_cast<const float4*>(p);
  u.q[1] = *reinterpret_cast<const float4*>(p + 16);
  return u.v;
}

// ---------------------------------------------------------------------------
// Elementwise helpers
// ---------------------------------------------------------------------------
__global__ void cvt_f32_bf16_kernel(const float* __restrict__ src,
                                    u16* __restrict__ dst, int n) {
  int i = blockIdx.x * blockDim.x + threadIdx.x;
  if (i < n) dst[i] = f2bf(src[i]);
}

// v (T x D) f32 -> vT (D x T) bf16
__global__ void transpose_cvt_kernel(const float* __restrict__ src,
                                     u16* __restrict__ dst) {
  int i = blockIdx.x * blockDim.x + threadIdx.x;
  if (i < TDIM * DMODEL) {
    int t = i / DMODEL, d = i % DMODEL;
    dst[(size_t)d * TDIM + t] = f2bf(src[i]);
  }
}

// betaT[h][t] = 2*sigmoid(pre[t][h] + bias[h])
__global__ void beta_kernel(const float* __restrict__ pre,
                            const float* __restrict__ bias,
                            float* __restrict__ betaT) {
  int i = blockIdx.x * blockDim.x + threadIdx.x;
  if (i < TDIM * HRh) {
    int t = i / HRh, c = i % HRh;
    float x = pre[(size_t)t * HRh + c] + bias[c];
    betaT[(size_t)c * TDIM + t] = 2.f / (1.f + __expf(-x));
  }
}

// Causal depthwise conv(K=3) + SiLU + rotation + L2-normalize -> wh bf16 [HR][T][HD]
__global__ void conv_rot_norm_kernel(const float* __restrict__ wraw,
                                     const float* __restrict__ conv_w,
                                     const float* __restrict__ omega,
                                     const float* __restrict__ phi,
                                     u16* __restrict__ wh) {
  int t  = blockIdx.x;
  int hr = blockIdx.y;
  int d  = threadIdx.x;           // 64 threads
  int hk = hr >> 1, r = hr & 1;
  int cA = (hk * 4 + 2 * r) * HDh + d;
  int cB = cA + HDh;
  float a = 0.f, bb = 0.f;
  #pragma unroll
  for (int i = 0; i < 3; ++i) {
    int ts = t + i - 2;
    if (ts >= 0) {
      a  += wraw[(size_t)ts * OUTWc + cA] * conv_w[cA * 3 + i];
      bb += wraw[(size_t)ts * OUTWc + cB] * conv_w[cB * 3 + i];
    }
  }
  a  = a  / (1.f + __expf(-a));    // SiLU
  bb = bb / (1.f + __expf(-bb));
  float th = omega[r] * (float)t + phi[r];
  float wv = a * __cosf(th) + bb * __sinf(th);
  __shared__ float sq[64];
  sq[d] = wv * wv;
  __syncthreads();
  for (int s = 32; s > 0; s >>= 1) {
    if (d < s) sq[d] += sq[d + s];
    __syncthreads();
  }
  float inv = rsqrtf(sq[0] + 1e-6f);
  wh[((size_t)hr * TDIM + t) * HDh + d] = f2bf(wv * inv);
}

// ---------------------------------------------------------------------------
// Generic C(f32) = A(bf16, MxK) * B(bf16, NxK)^T ; 4 waves per block, 64x64 tile
// ---------------------------------------------------------------------------
__global__ void gemm_nt_kernel(const u16* __restrict__ A, const u16* __restrict__ B,
                               float* __restrict__ C, int M, int N, int K,
                               int lda, int ldb, int ldc) {
  int wave = threadIdx.x >> 5;
  int row = blockIdx.y * 64 + (wave >> 1) * 32;
  int col = blockIdx.x * 64 + (wave & 1) * 32;
  if (row >= M || col >= N) return;   // wave-uniform
  v8f c00 = ZERO8, c01 = ZERO8, c10 = ZERO8, c11 = ZERO8;
  for (int k = 0; k < K; k += 32) {
    __builtin_prefetch(A + (size_t)row * lda + k + 128, 0, 3);
    __builtin_prefetch(B + (size_t)col * ldb + k + 128, 0, 3);
    v16bf a0 = frag_a(A, lda, row, k);
    v16bf a1 = frag_a(A, lda, row + 16, k);
    v16bf b0 = frag_b(B, ldb, col, k);
    v16bf b1 = frag_b(B, ldb, col + 16, k);
    c00 = WMMA(a0, b0, c00);
    c01 = WMMA(a0, b1, c01);
    c10 = WMMA(a1, b0, c10);
    c11 = WMMA(a1, b1, c11);
  }
  int lane = threadIdx.x & 31;
  int tn = lane & 15, mh = (lane >> 4) << 3;
  #pragma unroll
  for (int r = 0; r < 8; ++r) {
    C[(size_t)(row + mh + r)      * ldc + col + tn]      = c00[r];
    C[(size_t)(row + mh + r)      * ldc + col + 16 + tn] = c01[r];
    C[(size_t)(row + 16 + mh + r) * ldc + col + tn]      = c10[r];
    C[(size_t)(row + 16 + mh + r) * ldc + col + 16 + tn] = c11[r];
  }
}

// ---------------------------------------------------------------------------
// Mw[t][s] = (t>s) ? beta[t] * <wh_t, wh_s> : 0   (bf16)
// rhs[t][s]= (t>s) ? beta[t] * <wh_t, kh_s> : 0   (f32, lands in Bf buffer)
// ---------------------------------------------------------------------------
__device__ __forceinline__ void mwrhs_store(u16* Mwh, float* Rh, const float* bet,
                                            int row, int col, v8f cm, v8f cr) {
  int lane = threadIdx.x & 31;
  int n = col + (lane & 15);
  int m0 = row + ((lane >> 4) << 3);
  #pragma unroll
  for (int r = 0; r < 8; ++r) {
    int t = m0 + r;
    float b = (t > n) ? bet[t] : 0.f;
    Mwh[(size_t)t * TDIM + n] = f2bf(cm[r] * b);
    Rh[(size_t)t * TDIM + n]  = cr[r] * b;
  }
}

__global__ void mw_rhs_kernel(const u16* __restrict__ wh, const u16* __restrict__ kbf,
                              const float* __restrict__ betaT,
                              u16* __restrict__ Mw, float* __restrict__ Rhs) {
  int h = blockIdx.z;
  const u16* W  = wh  + (size_t)h * TDIM * HDh;          // ld HDh
  const u16* Kh = kbf + (size_t)(h >> 1) * HDh;          // ld DMODEL
  u16*   Mwh = Mw  + (size_t)h * TDIM * TDIM;
  float* Rh  = Rhs + (size_t)h * TDIM * TDIM;
  const float* bet = betaT + (size_t)h * TDIM;
  int wave = threadIdx.x >> 5;
  int row = blockIdx.y * 64 + (wave >> 1) * 32;  // t
  int col = blockIdx.x * 64 + (wave & 1) * 32;   // s
  v8f m00 = ZERO8, m01 = ZERO8, m10 = ZERO8, m11 = ZERO8;
  v8f r00 = ZERO8, r01 = ZERO8, r10 = ZERO8, r11 = ZERO8;
  if (row + 31 > col) {                          // wave-uniform: below-diag work only
    for (int k = 0; k < HDh; k += 32) {
      v16bf a0 = frag_a(W, HDh, row, k);
      v16bf a1 = frag_a(W, HDh, row + 16, k);
      v16bf wb0 = frag_b(W, HDh, col, k);
      v16bf wb1 = frag_b(W, HDh, col + 16, k);
      v16bf kb0 = frag_b(Kh, DMODEL, col, k);
      v16bf kb1 = frag_b(Kh, DMODEL, col + 16, k);
      m00 = WMMA(a0, wb0, m00); m01 = WMMA(a0, wb1, m01);
      m10 = WMMA(a1, wb0, m10); m11 = WMMA(a1, wb1, m11);
      r00 = WMMA(a0, kb0, r00); r01 = WMMA(a0, kb1, r01);
      r10 = WMMA(a1, kb0, r10); r11 = WMMA(a1, kb1, r11);
    }
  }
  mwrhs_store(Mwh, Rh, bet, row,      col,      m00, r00);
  mwrhs_store(Mwh, Rh, bet, row,      col + 16, m01, r01);
  mwrhs_store(Mwh, Rh, bet, row + 16, col,      m10, r10);
  mwrhs_store(Mwh, Rh, bet, row + 16, col + 16, m11, r11);
}

// ---------------------------------------------------------------------------
// Blocked forward substitution step ib: rows [ib*64, ib*64+64).
//  P = Mw[ib, 0:ib*64] @ Bsolved          (WMMA over transposed bf16 copy BT)
//  tmp = rhs - P ; solve unit-lower 64x64 diag block in LDS ; write Bf + BT.
// ---------------------------------------------------------------------------
__global__ void solve_step_kernel(int ib, const u16* __restrict__ Mw,
                                  u16* __restrict__ BT, float* __restrict__ Bf) {
  int h = blockIdx.y;
  const u16* Mwh = Mw + (size_t)h * TDIM * TDIM;
  u16*   BTh = BT + (size_t)h * TDIM * TDIM;
  float* Bh  = Bf + (size_t)h * TDIM * TDIM;
  int col0 = blockIdx.x * 64;
  int wave = threadIdx.x >> 5;
  int wr = (wave >> 1) * 32, wc = (wave & 1) * 32;
  int base = ib * 64;
  v8f p00 = ZERO8, p01 = ZERO8, p10 = ZERO8, p11 = ZERO8;
  for (int k = 0; k < base; k += 32) {
    v16bf a0 = frag_a(Mwh, TDIM, base + wr, k);
    v16bf a1 = frag_a(Mwh, TDIM, base + wr + 16, k);
    v16bf b0 = frag_b(BTh, TDIM, col0 + wc, k);
    v16bf b1 = frag_b(BTh, TDIM, col0 + wc + 16, k);
    p00 = WMMA(a0, b0, p00); p01 = WMMA(a0, b1, p01);
    p10 = WMMA(a1, b0, p10); p11 = WMMA(a1, b1, p11);
  }
  __shared__ float sh_t[64][65];
  __shared__ float sh_d[64][64];
  int lane = threadIdx.x & 31;
  int tn = lane & 15, mh = (lane >> 4) << 3;
  #pragma unroll
  for (int r = 0; r < 8; ++r) {
    int lm = wr + mh + r;
    sh_t[lm][wc + tn]      = Bh[(size_t)(base + lm) * TDIM + col0 + wc + tn]      - p00[r];
    sh_t[lm][wc + 16 + tn] = Bh[(size_t)(base + lm) * TDIM + col0 + wc + 16 + tn] - p01[r];
    int lm2 = wr + 16 + mh + r;
    sh_t[lm2][wc + tn]      = Bh[(size_t)(base + lm2) * TDIM + col0 + wc + tn]      - p10[r];
    sh_t[lm2][wc + 16 + tn] = Bh[(size_t)(base + lm2) * TDIM + col0 + wc + 16 + tn] - p11[r];
  }
  for (int e = threadIdx.x; e < 64 * 64; e += 128) {
    int rr = e >> 6, ss = e & 63;
    sh_d[rr][ss] = bf2f(Mwh[(size_t)(base + rr) * TDIM + base + ss]);
  }
  __syncthreads();
  if (threadIdx.x < 64) {                 // per-column forward substitution
    int c = threadIdx.x;
    for (int r = 0; r < 64; ++r) {
      float x = sh_t[r][c];
      for (int s = 0; s < r; ++s) x -= sh_d[r][s] * sh_t[s][c];
      sh_t[r][c] = x;
    }
  }
  __syncthreads();
  for (int e = threadIdx.x; e < 64 * 64; e += 128) {
    int rr = e >> 6, cc = e & 63;
    float x = sh_t[rr][cc];
    Bh[(size_t)(base + rr) * TDIM + col0 + cc] = x;
    BTh[(size_t)(col0 + cc) * TDIM + base + rr] = f2bf(x);
  }
}

// ---------------------------------------------------------------------------
// qk[i][j] (f32 -> logits buffer) and qw[i][t] = (i>=t)?<q_i,w_t>:0 (bf16)
// ---------------------------------------------------------------------------
__global__ void qkqw_kernel(const u16* __restrict__ q_bf, const u16* __restrict__ k_bf,
                            const u16* __restrict__ wh,
                            float* __restrict__ qkbuf, u16* __restrict__ qw_bf) {
  int h = blockIdx.z;
  int hv = h >> 1;
  const u16* Q  = q_bf + (size_t)hv * HDh;               // ld DMODEL
  const u16* Kh = k_bf + (size_t)hv * HDh;               // ld DMODEL
  const u16* W  = wh + (size_t)h * TDIM * HDh;           // ld HDh
  float* qk = qkbuf + (size_t)h * TDIM * TDIM;
  u16*   qw = qw_bf + (size_t)h * TDIM * TDIM;
  int wave = threadIdx.x >> 5;
  int row = blockIdx.y * 64 + (wave >> 1) * 32;  // i
  int col = blockIdx.x * 64 + (wave & 1) * 32;   // j / t
  v8f k00 = ZERO8, k01 = ZERO8, k10 = ZERO8, k11 = ZERO8;
  v8f w00 = ZERO8, w01 = ZERO8, w10 = ZERO8, w11 = ZERO8;
  for (int k = 0; k < HDh; k += 32) {
    v16bf a0  = frag_a(Q, DMODEL, row, k);
    v16bf a1  = frag_a(Q, DMODEL, row + 16, k);
    v16bf kb0 = frag_b(Kh, DMODEL, col, k);
    v16bf kb1 = frag_b(Kh, DMODEL, col + 16, k);
    v16bf wb0 = frag_b(W, HDh, col, k);
    v16bf wb1 = frag_b(W, HDh, col + 16, k);
    k00 = WMMA(a0, kb0, k00); k01 = WMMA(a0, kb1, k01);
    k10 = WMMA(a1, kb0, k10); k11 = WMMA(a1, kb1, k11);
    w00 = WMMA(a0, wb0, w00); w01 = WMMA(a0, wb1, w01);
    w10 = WMMA(a1, wb0, w10); w11 = WMMA(a1, wb1, w11);
  }
  int lane = threadIdx.x & 31;
  int tn = lane & 15, mh = (lane >> 4) << 3;
  #pragma unroll
  for (int r = 0; r < 8; ++r) {
    int i0 = row + mh + r, i1 = row + 16 + mh + r;
    int n0 = col + tn, n1 = col + 16 + tn;
    qk[(size_t)i0 * TDIM + n0] = k00[r];
    qk[(size_t)i0 * TDIM + n1] = k01[r];
    qk[(size_t)i1 * TDIM + n0] = k10[r];
    qk[(size_t)i1 * TDIM + n1] = k11[r];
    qw[(size_t)i0 * TDIM + n0] = (i0 >= n0) ? f2bf(w00[r]) : (u16)0;
    qw[(size_t)i0 * TDIM + n1] = (i0 >= n1) ? f2bf(w01[r]) : (u16)0;
    qw[(size_t)i1 * TDIM + n0] = (i1 >= n0) ? f2bf(w10[r]) : (u16)0;
    qw[(size_t)i1 * TDIM + n1] = (i1 >= n1) ? f2bf(w11[r]) : (u16)0;
  }
}

// logits[i][j] = (i>=j) ? (qk - sum_t qw[i][t]*Bmat[t][j]) * SCALE : -1e30
__global__ void logits_kernel(const u16* __restrict__ qw_bf, const u16* __restrict__ BT,
                              float* __restrict__ lg) {
  int h = blockIdx.z;
  const u16* A = qw_bf + (size_t)h * TDIM * TDIM;
  const u16* B = BT + (size_t)h * TDIM * TDIM;
  float* L = lg + (size_t)h * TDIM * TDIM;
  int wave = threadIdx.x >> 5;
  int row = blockIdx.y * 64 + (wave >> 1) * 32;
  int col = blockIdx.x * 64 + (wave & 1) * 32;
  v8f c00 = ZERO8, c01 = ZERO8, c10 = ZERO8, c11 = ZERO8;
  for (int k = 0; k < TDIM; k += 32) {
    v16bf a0 = frag_a(A, TDIM, row, k);
    v16bf a1 = frag_a(A, TDIM, row + 16, k);
    v16bf b0 = frag_b(B, TDIM, col, k);
    v16bf b1 = frag_b(B, TDIM, col + 16, k);
    c00 = WMMA(a0, b0, c00); c01 = WMMA(a0, b1, c01);
    c10 = WMMA(a1, b0, c10); c11 = WMMA(a1, b1, c11);
  }
  int lane = threadIdx.x & 31;
  int tn = lane & 15, mh = (lane >> 4) << 3;
  #pragma unroll
  for (int r = 0; r < 8; ++r) {
    int i0 = row + mh + r, i1 = row + 16 + mh + r;
    int n0 = col + tn, n1 = col + 16 + tn;
    float v00 = (L[(size_t)i0 * TDIM + n0] - c00[r]) * SCALEc;
    float v01 = (L[(size_t)i0 * TDIM + n1] - c01[r]) * SCALEc;
    float v10 = (L[(size_t)i1 * TDIM + n0] - c10[r]) * SCALEc;
    float v11 = (L[(size_t)i1 * TDIM + n1] - c11[r]) * SCALEc;
    L[(size_t)i0 * TDIM + n0] = (i0 >= n0) ? v00 : -1e30f;
    L[(size_t)i0 * TDIM + n1] = (i0 >= n1) ? v01 : -1e30f;
    L[(size_t)i1 * TDIM + n0] = (i1 >= n0) ? v10 : -1e30f;
    L[(size_t)i1 * TDIM + n1] = (i1 >= n1) ? v11 : -1e30f;
  }
}

// Row softmax, output bf16 probabilities (reuses qw buffer).
__global__ void softmax_kernel(const float* __restrict__ lg, u16* __restrict__ p_bf) {
  int h = blockIdx.y, i = blockIdx.x;
  const float* row = lg + ((size_t)h * TDIM + i) * TDIM;
  u16* out = p_bf + ((size_t)h * TDIM + i) * TDIM;
  __shared__ float red[256];
  float m = -3.4e38f;
  for (int j = threadIdx.x; j < TDIM; j += 256) m = fmaxf(m, row[j]);
  red[threadIdx.x] = m;
  __syncthreads();
  for (int s = 128; s > 0; s >>= 1) {
    if (threadIdx.x < s) red[threadIdx.x] = fmaxf(red[threadIdx.x], red[threadIdx.x + s]);
    __syncthreads();
  }
  m = red[0];
  __syncthreads();
  float sum = 0.f;
  for (int j = threadIdx.x; j < TDIM; j += 256) sum += __expf(row[j] - m);
  red[threadIdx.x] = sum;
  __syncthreads();
  for (int s = 128; s > 0; s >>= 1) {
    if (threadIdx.x < s) red[threadIdx.x] += red[threadIdx.x + s];
    __syncthreads();
  }
  float inv = 1.f / red[0];
  for (int j = threadIdx.x; j < TDIM; j += 256)
    out[j] = f2bf(__expf(row[j] - m) * inv);
}

// o[i][h*64+d] = sum_j p[i][j] * v[j][d]  (bf16 out for the final projection)
__global__ void pv_kernel(const u16* __restrict__ p_bf, const u16* __restrict__ vT,
                          u16* __restrict__ o_bf) {
  int h = blockIdx.z;
  const u16* A = p_bf + (size_t)h * TDIM * TDIM;           // ld TDIM
  const u16* B = vT + (size_t)(h >> 1) * HDh * TDIM;       // ld TDIM, rows = d
  int wave = threadIdx.x >> 5;
  int row = blockIdx.y * 64 + (wave >> 1) * 32;
  int col = (wave & 1) * 32;                               // N = 64
  v8f c00 = ZERO8, c01 = ZERO8, c10 = ZERO8, c11 = ZERO8;
  for (int k = 0; k < TDIM; k += 32) {
    v16bf a0 = frag_a(A, TDIM, row, k);
    v16bf a1 = frag_a(A, TDIM, row + 16, k);
    v16bf b0 = frag_b(B, TDIM, col, k);
    v16bf b1 = frag_b(B, TDIM, col + 16, k);
    c00 = WMMA(a0, b0, c00); c01 = WMMA(a0, b1, c01);
    c10 = WMMA(a1, b0, c10); c11 = WMMA(a1, b1, c11);
  }
  int lane = threadIdx.x & 31;
  int tn = lane & 15, mh = (lane >> 4) << 3;
  #pragma unroll
  for (int r = 0; r < 8; ++r) {
    int i0 = row + mh + r, i1 = row + 16 + mh + r;
    o_bf[(size_t)i0 * 2048 + h * 64 + col + tn]      = f2bf(c00[r]);
    o_bf[(size_t)i0 * 2048 + h * 64 + col + 16 + tn] = f2bf(c01[r]);
    o_bf[(size_t)i1 * 2048 + h * 64 + col + tn]      = f2bf(c10[r]);
    o_bf[(size_t)i1 * 2048 + h * 64 + col + 16 + tn] = f2bf(c11[r]);
  }
}

// ---------------------------------------------------------------------------
// Host orchestration
// ---------------------------------------------------------------------------
extern "C" void kernel_launch(void* const* d_in, const int* in_sizes, int n_in,
                              void* d_out, int out_size, void* d_ws, size_t ws_size,
                              hipStream_t stream) {
  (void)in_sizes; (void)n_in; (void)out_size; (void)ws_size;
  const float* hs     = (const float*)d_in[0];
  const float* q_w    = (const float*)d_in[1];
  const float* k_w    = (const float*)d_in[2];
  const float* v_w    = (const float*)d_in[3];
  const float* w1     = (const float*)d_in[4];
  const float* w2     = (const float*)d_in[5];
  const float* conv_w = (const float*)d_in[6];
  const float* bt_w   = (const float*)d_in[7];
  const float* bt_b   = (const float*)d_in[8];
  const float* o_w    = (const float*)d_in[9];
  const float* omega  = (const float*)d_in[10];
  const float* phi    = (const float*)d_in[11];

  char* wsb = (char*)d_ws;
  size_t off = 0;
  auto alloc = [&](size_t bytes) -> void* {
    void* p = wsb + off;
    off = (off + bytes + 255) & ~(size_t)255;
    return p;
  };
  const size_t T = 1024, Dm = 1024, HR = 32, OW = 4096;

  u16*   hs_bf  = (u16*)alloc(T * Dm * 2);
  u16*   qW_bf  = (u16*)alloc(Dm * Dm * 2);
  u16*   kW_bf  = (u16*)alloc(Dm * Dm * 2);
  u16*   vW_bf  = (u16*)alloc(Dm * Dm * 2);
  u16*   w1_bf  = (u16*)alloc(32 * Dm * 2);
  u16*   w2_bf  = (u16*)alloc(OW * 32 * 2);
  u16*   btW_bf = (u16*)alloc(32 * Dm * 2);
  u16*   oW_bf  = (u16*)alloc(Dm * 2 * Dm * 2);
  float* q_f    = (float*)alloc(T * Dm * 4);
  float* k_f    = (float*)alloc(T * Dm * 4);
  float* v_f    = (float*)alloc(T * Dm * 4);
  float* t1_f   = (float*)alloc(T * 32 * 4);
  float* bpre   = (float*)alloc(T * 32 * 4);
  float* wraw   = (float*)alloc(T * OW * 4);
  u16*   q_bf   = (u16*)alloc(T * Dm * 2);
  u16*   k_bf   = (u16*)alloc(T * Dm * 2);
  u16*   vT_bf  = (u16*)alloc(Dm * T * 2);
  u16*   t1_bf  = (u16*)alloc(T * 32 * 2);
  u16*   wh_bf  = (u16*)alloc(HR * T * 64 * 2);
  float* betaT  = (float*)alloc(HR * T * 4);
  u16*   Mw_bf  = (u16*)alloc((size_t)HR * T * T * 2);
  float* Bf     = (float*)alloc((size_t)HR * T * T * 4);
  u16*   BT_bf  = (u16*)alloc((size_t)HR * T * T * 2);
  u16*   qw_bf  = (u16*)alloc((size_t)HR * T * T * 2);   // reused as p_bf
  float* lg     = (float*)alloc((size_t)HR * T * T * 4);
  u16*   o_bf   = (u16*)alloc(T * 2048 * 2);

  dim3 blk(128);
  auto cvt = [&](const float* s, u16* d, int n) {
    cvt_f32_bf16_kernel<<<dim3((n + 255) / 256), dim3(256), 0, stream>>>(s, d, n);
  };

  // f32 -> bf16 operand prep
  cvt(hs, hs_bf, (int)(T * Dm));
  cvt(q_w, qW_bf, (int)(Dm * Dm));
  cvt(k_w, kW_bf, (int)(Dm * Dm));
  cvt(v_w, vW_bf, (int)(Dm * Dm));
  cvt(w1, w1_bf, (int)(32 * Dm));
  cvt(w2, w2_bf, (int)(OW * 32));
  cvt(bt_w, btW_bf, (int)(32 * Dm));
  cvt(o_w, oW_bf, (int)(Dm * 2 * Dm));

  // Projections
  gemm_nt_kernel<<<dim3(16, 16), blk, 0, stream>>>(hs_bf, qW_bf, q_f, 1024, 1024, 1024, 1024, 1024, 1024);
  gemm_nt_kernel<<<dim3(16, 16), blk, 0, stream>>>(hs_bf, kW_bf, k_f, 1024, 1024, 1024, 1024, 1024, 1024);
  gemm_nt_kernel<<<dim3(16, 16), blk, 0, stream>>>(hs_bf, vW_bf, v_f, 1024, 1024, 1024, 1024, 1024, 1024);
  gemm_nt_kernel<<<dim3(1, 16), blk, 0, stream>>>(hs_bf, w1_bf, t1_f, 1024, 32, 1024, 1024, 1024, 32);
  gemm_nt_kernel<<<dim3(1, 16), blk, 0, stream>>>(hs_bf, btW_bf, bpre, 1024, 32, 1024, 1024, 1024, 32);

  cvt(q_f, q_bf, (int)(T * Dm));
  cvt(k_f, k_bf, (int)(T * Dm));
  cvt(t1_f, t1_bf, (int)(T * 32));
  transpose_cvt_kernel<<<dim3((int)((T * Dm + 255) / 256)), dim3(256), 0, stream>>>(v_f, vT_bf);

  // Low-rank w path, conv/rot/norm, beta
  gemm_nt_kernel<<<dim3(64, 16), blk, 0, stream>>>(t1_bf, w2_bf, wraw, 1024, 4096, 32, 32, 32, 4096);
  conv_rot_norm_kernel<<<dim3(1024, 32), dim3(64), 0, stream>>>(wraw, conv_w, omega, phi, wh_bf);
  beta_kernel<<<dim3((int)((T * 32 + 255) / 256)), dim3(256), 0, stream>>>(bpre, bt_b, betaT);

  // Mw / rhs, then 16-step blocked triangular solve
  mw_rhs_kernel<<<dim3(16, 16, 32), blk, 0, stream>>>(wh_bf, k_bf, betaT, Mw_bf, Bf);
  for (int ib = 0; ib < 16; ++ib)
    solve_step_kernel<<<dim3(16, 32), blk, 0, stream>>>(ib, Mw_bf, BT_bf, Bf);

  // Attention core
  qkqw_kernel<<<dim3(16, 16, 32), blk, 0, stream>>>(q_bf, k_bf, wh_bf, lg, qw_bf);
  logits_kernel<<<dim3(16, 16, 32), blk, 0, stream>>>(qw_bf, BT_bf, lg);
  softmax_kernel<<<dim3(1024, 32), dim3(256), 0, stream>>>(lg, qw_bf);   // qw_bf -> p
  pv_kernel<<<dim3(1, 16, 32), blk, 0, stream>>>(qw_bf, vT_bf, o_bf);

  // Final projection -> d_out (f32)
  gemm_nt_kernel<<<dim3(16, 16), blk, 0, stream>>>(o_bf, oW_bf, (float*)d_out,
                                                   1024, 1024, 2048, 2048, 2048, 1024);
}
